// lenetModel_14886356648869
// MI455X (gfx1250) — matrix operand order repaired and checked by reading, not compiled, not run on hardware
//
#include <hip/hip_runtime.h>

typedef __attribute__((ext_vector_type(16))) _Float16 v16h;
typedef __attribute__((ext_vector_type(8)))  _Float16 v8h;
typedef __attribute__((ext_vector_type(8)))  float    v8f;

static __device__ __forceinline__ v8f wmma_f16(v16h a, v16h b, v8f c) {
  // D = A(16x32 f16) * B(32x16 f16) + C(16x16 f32)
  return __builtin_amdgcn_wmma_f32_16x16x32_f16(false, a, false, b, (short)0, c, false, false);
}

#define INVOFF (1 << 20)   // invalid-offset sentinel; addr clamps to zero pad

// Permute K within each 32-chunk (swap the middle two 8-groups) so that a
// lane's A-fragment (K = 32*s + 8*hi + (j<8?j:j+8)) becomes 16 contiguous
// halves at offset 32*s + 16*hi -> one v16h LDS load, no packing movs.
static __device__ __forceinline__ int permK(int k) {
  int g  = (k >> 3) & 3;
  int pg = ((g & 1) << 1) | (g >> 1);
  return (k & ~31) | (pg << 3) | (k & 7);
}

// LeNet C3 sparse connectivity
__device__ const int kGroups[16][6] = {
  {0,1,2,-1,-1,-1},{1,2,3,-1,-1,-1},{2,3,4,-1,-1,-1},{3,4,5,-1,-1,-1},
  {0,4,5,-1,-1,-1},{0,1,5,-1,-1,-1},
  {0,1,2,3,-1,-1},{1,2,3,4,-1,-1},{2,3,4,5,-1,-1},{0,3,4,5,-1,-1},
  {0,1,4,5,-1,-1},{0,1,2,5,-1,-1},
  {0,1,3,4,-1,-1},{1,2,4,5,-1,-1},{0,2,3,5,-1,-1},
  {0,1,2,3,4,5}};
__device__ const int kGroupN[16] = {3,3,3,3,3,3,4,4,4,4,4,4,4,4,4,6};

// ---------------------------------------------------------------------------
// Prep: pack all weights into exact WMMA B-fragment lane order (f16), build
// the dense C3 weight from the sparse table, concat the 16-entry conv2 bias.
// B-frag element j of lane l within a K-step: K = 32*step + 8*(l>>4) + (j<8?j:j+8),
// N = l & 15.  Fragment storage: [(fragIdx*32 + lane)*16 + j].
// ---------------------------------------------------------------------------
__global__ __launch_bounds__(256) void prep_kernel(
    const float* __restrict__ c1_w,
    const float* __restrict__ w1, const float* __restrict__ w2c,
    const float* __restrict__ w3c, const float* __restrict__ w4c,
    const float* __restrict__ b1, const float* __restrict__ b2c,
    const float* __restrict__ b3c, const float* __restrict__ b4c,
    const float* __restrict__ conv3_w, const float* __restrict__ ll1_w,
    _Float16* __restrict__ c1pack, _Float16* __restrict__ w2pack,
    _Float16* __restrict__ w3pack, _Float16* __restrict__ w1pack,
    float* __restrict__ bias16)
{
  const int A_ = 3 * 512;        // conv1: 3 K-steps, 1 N-tile
  const int B_ = 5 * 512;        // conv2: 5 K-steps, 1 N-tile
  const int C_ = 13 * 8 * 512;   // conv3: 13 K-steps, 8 N-tiles
  const int D_ = 4 * 6 * 512;    // fc1:   4 K-steps, 6 N-tiles
  int gid = blockIdx.x * 256 + threadIdx.x;

  if (gid < A_) {
    int e = gid, lane = (e >> 4) & 31, j = e & 15, step = e >> 9;
    int nidx = lane & 15;
    int k = step * 32 + ((lane >> 4) << 3) + (j < 8 ? j : j + 8);
    float v = (nidx < 6 && k < 75) ? c1_w[nidx * 75 + k] : 0.f;
    c1pack[e] = (_Float16)v;
  } else if (gid < A_ + B_) {
    int e = gid - A_, lane = (e >> 4) & 31, j = e & 15, step = e >> 9;
    int o = lane & 15;
    int k = step * 32 + ((lane >> 4) << 3) + (j < 8 ? j : j + 8);
    float v = 0.f;
    if (k < 150) {
      int ci = k / 25, r = k % 25;
      const float* part;
      if (o < 6)       part = w1  + o * 75;
      else if (o < 12) part = w2c + (o - 6) * 100;
      else if (o < 15) part = w3c + (o - 12) * 100;
      else             part = w4c;
      int cnt = kGroupN[o];
      for (int g = 0; g < cnt; ++g)
        if (kGroups[o][g] == ci) { v = part[g * 25 + r]; break; }
    }
    w2pack[e] = (_Float16)v;
  } else if (gid < A_ + B_ + C_) {
    int e = gid - A_ - B_, lane = (e >> 4) & 31, j = e & 15, fi = e >> 9;
    int ntile = fi & 7, step = fi >> 3;
    int nidx = ntile * 16 + (lane & 15);
    int k = step * 32 + ((lane >> 4) << 3) + (j < 8 ? j : j + 8);
    float v = (nidx < 120 && k < 400) ? conv3_w[nidx * 400 + k] : 0.f;
    w3pack[e] = (_Float16)v;
  } else if (gid < A_ + B_ + C_ + D_) {
    int e = gid - A_ - B_ - C_, lane = (e >> 4) & 31, j = e & 15, fi = e >> 9;
    int ntile = fi % 6, step = fi / 6;
    int nidx = ntile * 16 + (lane & 15);
    int k = step * 32 + ((lane >> 4) << 3) + (j < 8 ? j : j + 8);
    float v = (nidx < 84 && k < 120) ? ll1_w[nidx * 120 + k] : 0.f;
    w1pack[e] = (_Float16)v;
  } else if (gid < A_ + B_ + C_ + D_ + 16) {
    int o = gid - (A_ + B_ + C_ + D_);
    bias16[o] = (o < 6) ? b1[o] : (o < 12) ? b2c[o - 6] : (o < 15) ? b3c[o - 12] : b4c[0];
  }
}

// ---------------------------------------------------------------------------
// conv1 (WMMA implicit GEMM) + maxpool2 + per-block BN1 partial stats.
// One block per sample, 8 waves; 49 pixel-tiles of 16 over 28x28.
// A-gather uses clamped addresses into a zero pad slot: no EXEC-masked loads.
// ---------------------------------------------------------------------------
__global__ __launch_bounds__(256) void conv1_pool_kernel(
    const float* __restrict__ x, const float* __restrict__ c1_b,
    const _Float16* __restrict__ c1pack,
    float* __restrict__ pooled1, float* __restrict__ part1)
{
  __shared__ _Float16 sx[3 * 32 * 32 + 16];   // +pad, sx[3072] == 0
  __shared__ float sconv[6 * 28 * 28];
  __shared__ float red[256 * 12];
  const int n = blockIdx.x, tid = threadIdx.x;
  const float* xp = x + (size_t)n * 3072;
  __builtin_prefetch(xp + tid * 12, 0, 0);    // global_prefetch path
  for (int i = tid; i < 3072; i += 256) sx[i] = (_Float16)xp[i];
  if (tid == 0) sx[3072] = (_Float16)0.f;

  const int lane = tid & 31, wave = tid >> 5, hi = lane >> 4, lm = lane & 15;

  // A-fragment LDS offsets (constant per lane): K = 32*s + 8*hi + (j<8?j:j+8)
  int offs[3][16];
#pragma unroll
  for (int s = 0; s < 3; ++s)
#pragma unroll
    for (int j = 0; j < 16; ++j) {
      int k = 32 * s + 8 * hi + (j < 8 ? j : j + 8);
      offs[s][j] = (k < 75) ? ((k / 25) * 1024 + ((k % 25) / 5) * 32 + (k % 25) % 5)
                            : INVOFF;
    }
  v16h bf0 = *(const v16h*)(c1pack + (0 * 32 + lane) * 16);
  v16h bf1 = *(const v16h*)(c1pack + (1 * 32 + lane) * 16);
  v16h bf2 = *(const v16h*)(c1pack + (2 * 32 + lane) * 16);
  __syncthreads();

  for (int t = wave; t < 49; t += 8) {            // wave-uniform
    int m = t * 16 + lm;                          // output pixel, < 784
    int pb = (m / 28) * 32 + (m % 28);
    v8f c = {};
#pragma unroll
    for (int s = 0; s < 3; ++s) {
      v16h a = {};
#pragma unroll
      for (int j = 0; j < 16; ++j) {
        int addr = offs[s][j] + pb;
        addr = (addr < 3072) ? addr : 3072;       // v_min, clamp to zero pad
        a[j] = sx[addr];
      }
      v16h b = (s == 0) ? bf0 : (s == 1) ? bf1 : bf2;
      c = wmma_f16(a, b, c);
    }
    if (lm < 6) {
#pragma unroll
      for (int r = 0; r < 8; ++r)
        sconv[lm * 784 + t * 16 + r + 8 * hi] = c[r];
    }
  }
  __syncthreads();

  // maxpool 2x2 + bias, write pooled1; per-channel loop keeps channel indices
  // compile-time constant (no predicated accumulate chains)
  float s0[6] = {0, 0, 0, 0, 0, 0}, s1[6] = {0, 0, 0, 0, 0, 0};
#pragma unroll
  for (int cc = 0; cc < 6; ++cc) {
    float bb = c1_b[cc];
    for (int p = tid; p < 196; p += 256) {
      int py = p / 14, px = p % 14;
      int b0 = cc * 784 + py * 56 + px * 2;
      float v = fmaxf(fmaxf(sconv[b0], sconv[b0 + 1]),
                      fmaxf(sconv[b0 + 28], sconv[b0 + 29])) + bb;
      pooled1[(size_t)n * 1176 + cc * 196 + p] = v;
      s0[cc] += v; s1[cc] += v * v;
    }
  }
#pragma unroll
  for (int cc = 0; cc < 6; ++cc) { red[tid * 12 + cc * 2] = s0[cc]; red[tid * 12 + cc * 2 + 1] = s1[cc]; }
  __syncthreads();
  for (int st = 128; st > 0; st >>= 1) {
    if (tid < st) {
#pragma unroll
      for (int q = 0; q < 12; ++q) red[tid * 12 + q] += red[(tid + st) * 12 + q];
    }
    __syncthreads();
  }
  if (tid < 12) part1[(size_t)n * 12 + tid] = red[tid];
}

// ---------------------------------------------------------------------------
// Deterministic stats reduce + BN finalize (scale/shift), grid = nch blocks.
// ---------------------------------------------------------------------------
__global__ __launch_bounds__(256) void reducefin_kernel(
    const float* __restrict__ part, int nblocks, int nch, float count,
    const float* __restrict__ g, const float* __restrict__ b,
    float* __restrict__ scale_out, float* __restrict__ shift_out)
{
  __shared__ float rs[256], rs2[256];
  const int ch = blockIdx.x, tid = threadIdx.x;
  float s = 0.f, s2 = 0.f;
  for (int i = tid; i < nblocks; i += 256) {
    s  += part[(size_t)i * (2 * nch) + ch * 2 + 0];
    s2 += part[(size_t)i * (2 * nch) + ch * 2 + 1];
  }
  rs[tid] = s; rs2[tid] = s2;
  __syncthreads();
  for (int st = 128; st > 0; st >>= 1) {
    if (tid < st) { rs[tid] += rs[tid + st]; rs2[tid] += rs2[tid + st]; }
    __syncthreads();
  }
  if (tid == 0) {
    float m = rs[0] / count;
    float v = rs2[0] / count - m * m;
    float inv = rsqrtf(v + 1e-5f);
    float sc = g[ch] * inv;
    scale_out[ch] = sc;
    shift_out[ch] = b[ch] - m * sc;
  }
}

// ---------------------------------------------------------------------------
// BN1+ReLU (inline) -> conv2 (WMMA, dense 16x150 weight) -> maxpool2 -> stats.
// One block per sample; waves 0..6 each own one 16-pixel tile (covers 112>=100).
// ---------------------------------------------------------------------------
__global__ __launch_bounds__(256) void conv2_pool_kernel(
    const float* __restrict__ pooled1, const float* __restrict__ stats, // [0..5]=scale1,[6..11]=shift1
    const _Float16* __restrict__ w2pack, const float* __restrict__ bias16,
    float* __restrict__ pooled2, float* __restrict__ part2)
{
  __shared__ _Float16 sIn[1176 + 8];   // 6 x 14 x 14 normalized f16, sIn[1176]==0
  __shared__ float sconv2[1600];       // 16 x 100
  __shared__ float red[256 * 32];
  const int n = blockIdx.x, tid = threadIdx.x;
  const float* sc1 = stats;
  const float* sh1 = stats + 6;
  for (int i = tid; i < 1176; i += 256) {
    int ch = i / 196;
    float v = fmaxf(pooled1[(size_t)n * 1176 + i] * sc1[ch] + sh1[ch], 0.f);
    sIn[i] = (_Float16)v;
  }
  if (tid == 0) sIn[1176] = (_Float16)0.f;
  __syncthreads();

  const int lane = tid & 31, wave = tid >> 5, hi = lane >> 4, lm = lane & 15;
  if (wave < 7) {                     // wave-uniform branch; EXEC full inside
    int offs[5][16];
#pragma unroll
    for (int s = 0; s < 5; ++s)
#pragma unroll
      for (int j = 0; j < 16; ++j) {
        int k = 32 * s + 8 * hi + (j < 8 ? j : j + 8);
        offs[s][j] = (k < 150) ? ((k / 25) * 196 + ((k % 25) / 5) * 14 + (k % 25) % 5)
                               : INVOFF;
      }
    int m = wave * 16 + lm;
    int pb = (m < 100) ? (m / 10) * 14 + (m % 10) : INVOFF;  // poison invalid rows
    v8f c = {};
#pragma unroll
    for (int s = 0; s < 5; ++s) {
      v16h bfr = *(const v16h*)(w2pack + (s * 32 + lane) * 16);
      v16h a = {};
#pragma unroll
      for (int j = 0; j < 16; ++j) {
        int addr = offs[s][j] + pb;
        addr = (addr < 1176) ? addr : 1176;       // clamp to zero pad
        a[j] = sIn[addr];
      }
      c = wmma_f16(a, bfr, c);
    }
#pragma unroll
    for (int r = 0; r < 8; ++r) {
      int pix = wave * 16 + r + 8 * hi;
      if (pix < 100) sconv2[lm * 100 + pix] = c[r];
    }
  }
  __syncthreads();

  float s0[16], s1[16];
#pragma unroll
  for (int cc = 0; cc < 16; ++cc) { s0[cc] = 0.f; s1[cc] = 0.f; }
  for (int i = tid; i < 400; i += 256) {
    int ch = i / 25, p = i % 25, py = p / 5, px = p % 5;
    int b0 = ch * 100 + py * 20 + px * 2;
    float v = fmaxf(fmaxf(sconv2[b0], sconv2[b0 + 1]),
                    fmaxf(sconv2[b0 + 10], sconv2[b0 + 11])) + bias16[ch];
    pooled2[(size_t)n * 400 + i] = v;
#pragma unroll
    for (int cc = 0; cc < 16; ++cc) {
      float vv = (cc == ch) ? v : 0.f;
      s0[cc] += vv; s1[cc] += vv * vv;
    }
  }
#pragma unroll
  for (int cc = 0; cc < 16; ++cc) { red[tid * 32 + cc * 2] = s0[cc]; red[tid * 32 + cc * 2 + 1] = s1[cc]; }
  __syncthreads();
  for (int st = 128; st > 0; st >>= 1) {
    if (tid < st) {
#pragma unroll
      for (int q = 0; q < 32; ++q) red[tid * 32 + q] += red[(tid + st) * 32 + q];
    }
    __syncthreads();
  }
  if (tid < 32) part2[(size_t)n * 32 + tid] = red[tid];
}

// ---------------------------------------------------------------------------
// BN2+ReLU inline -> conv3 as GEMM (Nx400)*(400x120) with WMMA + BN3 partials.
// Block = 16 samples, 8 waves = 8 N-tiles (120 -> 128), 13 K-steps (400 -> 416).
// Activations staged in permK order: A fragment = one contiguous v16h load.
// ---------------------------------------------------------------------------
__global__ __launch_bounds__(256) void conv3_kernel(
    const float* __restrict__ pooled2,
    const float* __restrict__ sc2, const float* __restrict__ sh2,
    const _Float16* __restrict__ w3pack, const float* __restrict__ conv3_b,
    float* __restrict__ conv3raw, float* __restrict__ part3)
{
  __shared__ _Float16 sa[16 * 416];
  __shared__ float pl[256 * 2];
  const int blk = blockIdx.x, tid = threadIdx.x;
  for (int i = tid; i < 6400; i += 256) {
    int sl = i / 400, k = i % 400, ch = k / 25;
    float v = fmaxf(pooled2[((size_t)blk * 16 + sl) * 400 + k] * sc2[ch] + sh2[ch], 0.f);
    sa[sl * 416 + permK(k)] = (_Float16)v;
  }
  { int sl = tid >> 4; sa[sl * 416 + permK(400 + (tid & 15))] = (_Float16)0.f; } // K pad
  __syncthreads();

  const int lane = tid & 31, w = tid >> 5, hi = lane >> 4, lm = lane & 15;
  v8f c = {};
  for (int s = 0; s < 13; ++s) {
    v16h b = *(const v16h*)(w3pack + ((size_t)(s * 8 + w) * 32 + lane) * 16);
    v16h a = *(const v16h*)(sa + lm * 416 + 32 * s + 16 * hi);
    c = wmma_f16(a, b, c);
  }
  int ch = w * 16 + lm;
  bool chv = ch < 120;
  float bias = chv ? conv3_b[ch] : 0.f;
  float ps = 0.f, ps2 = 0.f;
#pragma unroll
  for (int r = 0; r < 8; ++r) {
    int sidx = blk * 16 + r + 8 * hi;
    float v = c[r] + bias;
    if (chv) conv3raw[(size_t)sidx * 120 + ch] = v;
    float vv = chv ? v : 0.f;
    ps += vv; ps2 += vv * vv;
  }
  pl[tid * 2] = ps; pl[tid * 2 + 1] = ps2;
  __syncthreads();
  if (tid < 120) {
    int ww = tid / 16, nn = tid % 16;
    int l0 = ww * 32 + nn, l1 = l0 + 16;
    part3[(size_t)blk * 240 + tid * 2 + 0] = pl[l0 * 2] + pl[l1 * 2];
    part3[(size_t)blk * 240 + tid * 2 + 1] = pl[l0 * 2 + 1] + pl[l1 * 2 + 1];
  }
}

// ---------------------------------------------------------------------------
// BN3 inline -> fc1 (WMMA, 120->84) + ReLU -> fc2 (84->2) -> softmax.
// Block = 16 samples, 6 waves = 6 N-tiles (84 -> 96), 4 K-steps (120 -> 128).
// Activations staged in permK order: A fragment = one contiguous v16h load.
// ---------------------------------------------------------------------------
__global__ __launch_bounds__(192) void head_kernel(
    const float* __restrict__ conv3raw,
    const float* __restrict__ sc3, const float* __restrict__ sh3,
    const _Float16* __restrict__ w1pack, const float* __restrict__ ll1_b,
    const float* __restrict__ ll2_w, const float* __restrict__ ll2_b,
    float* __restrict__ out)
{
  __shared__ _Float16 sf[16 * 128];
  __shared__ float sh1[16 * 96];
  const int blk = blockIdx.x, tid = threadIdx.x;
  for (int i = tid; i < 1920; i += 192) {
    int sl = i / 120, k = i % 120;
    sf[sl * 128 + permK(k)] =
        (_Float16)(conv3raw[((size_t)blk * 16 + sl) * 120 + k] * sc3[k] + sh3[k]);
  }
  for (int i = tid; i < 128; i += 192) {
    int sl = i >> 3;
    sf[sl * 128 + permK(120 + (i & 7))] = (_Float16)0.f;
  }
  __syncthreads();

  const int lane = tid & 31, w = tid >> 5, hi = lane >> 4, lm = lane & 15;
  v8f c = {};
#pragma unroll
  for (int s = 0; s < 4; ++s) {
    v16h b = *(const v16h*)(w1pack + ((s * 6 + w) * 32 + lane) * 16);
    v16h a = *(const v16h*)(sf + lm * 128 + 32 * s + 16 * hi);
    c = wmma_f16(a, b, c);
  }
  int ch = w * 16 + lm;
  if (ch < 84) {
    float bb = ll1_b[ch];
#pragma unroll
    for (int r = 0; r < 8; ++r) {
      int sl = r + 8 * hi;
      sh1[sl * 96 + ch] = fmaxf(c[r] + bb, 0.f);
    }
  }
  __syncthreads();
  if (tid < 16) {
    float l0 = ll2_b[0], l1 = ll2_b[1];
    for (int j = 0; j < 84; ++j) {
      float h = sh1[tid * 96 + j];
      l0 += h * ll2_w[j];
      l1 += h * ll2_w[84 + j];
    }
    float mx = fmaxf(l0, l1);
    float e0 = expf(l0 - mx), e1 = expf(l1 - mx);
    float inv = 1.f / (e0 + e1);
    size_t o = ((size_t)blk * 16 + tid) * 2;
    out[o] = e0 * inv;
    out[o + 1] = e1 * inv;
  }
}

// ---------------------------------------------------------------------------
extern "C" void kernel_launch(void* const* d_in, const int* in_sizes, int n_in,
                              void* d_out, int out_size, void* d_ws, size_t ws_size,
                              hipStream_t stream)
{
  (void)n_in; (void)out_size; (void)ws_size;
  const float* x       = (const float*)d_in[0];
  const float* c1_w    = (const float*)d_in[1];
  const float* c1_b    = (const float*)d_in[2];
  const float* bn1_g   = (const float*)d_in[3];
  const float* bn1_b   = (const float*)d_in[4];
  const float* w1      = (const float*)d_in[5];
  const float* b1      = (const float*)d_in[6];
  const float* w2      = (const float*)d_in[7];
  const float* b2      = (const float*)d_in[8];
  const float* w3      = (const float*)d_in[9];
  const float* b3      = (const float*)d_in[10];
  const float* w4      = (const float*)d_in[11];
  const float* b4      = (const float*)d_in[12];
  const float* bn2_g   = (const float*)d_in[13];
  const float* bn2_b   = (const float*)d_in[14];
  const float* conv3_w = (const float*)d_in[15];
  const float* conv3_b = (const float*)d_in[16];
  const float* bn3_g   = (const float*)d_in[17];
  const float* bn3_b   = (const float*)d_in[18];
  const float* ll1_w   = (const float*)d_in[19];
  const float* ll1_b   = (const float*)d_in[20];
  const float* ll2_w   = (const float*)d_in[21];
  const float* ll2_b   = (const float*)d_in[22];

  const int N   = in_sizes[0] / 3072;   // 8192
  const int nbG = N / 16;               // 512 blocks for GEMM stages

  float* ws = (float*)d_ws;
  size_t off = 0;
  float* pooled1  = ws + off; off += (size_t)N * 1176;
  float* pooled2  = ws + off; off += (size_t)N * 400;
  float* conv3raw = ws + off; off += (size_t)N * 120;
  float* part1    = ws + off; off += (size_t)N * 12;
  float* part2    = ws + off; off += (size_t)N * 32;
  float* part3    = ws + off; off += (size_t)nbG * 240;
  float* stats    = ws + off; off += 512;
  // stats: [0..5]=scale1 [6..11]=shift1 [12..27]=scale2 [28..43]=shift2
  //        [44..163]=scale3 [164..283]=shift3 [284..299]=bias16
  _Float16* c1pack   = (_Float16*)(ws + off); off += 768;    // 1536 halfs
  _Float16* w2pack   = (_Float16*)(ws + off); off += 1280;   // 2560 halfs
  _Float16* w3pack   = (_Float16*)(ws + off); off += 26624;  // 53248 halfs
  _Float16* w1packfc = (_Float16*)(ws + off); off += 6144;   // 12288 halfs

  prep_kernel<<<273, 256, 0, stream>>>(c1_w, w1, w2, w3, w4, b1, b2, b3, b4,
                                       conv3_w, ll1_w,
                                       c1pack, w2pack, w3pack, w1packfc,
                                       stats + 284);
  conv1_pool_kernel<<<N, 256, 0, stream>>>(x, c1_b, c1pack, pooled1, part1);
  reducefin_kernel<<<6, 256, 0, stream>>>(part1, N, 6, (float)N * 196.f,
                                          bn1_g, bn1_b, stats + 0, stats + 6);
  conv2_pool_kernel<<<N, 256, 0, stream>>>(pooled1, stats, w2pack, stats + 284,
                                           pooled2, part2);
  reducefin_kernel<<<16, 256, 0, stream>>>(part2, N, 16, (float)N * 25.f,
                                           bn2_g, bn2_b, stats + 12, stats + 28);
  conv3_kernel<<<nbG, 256, 0, stream>>>(pooled2, stats + 12, stats + 28,
                                        w3pack, conv3_b, conv3raw, part3);
  reducefin_kernel<<<120, 256, 0, stream>>>(part3, nbG, 120, (float)N,
                                            bn3_g, bn3_b, stats + 44, stats + 164);
  head_kernel<<<nbG, 192, 0, stream>>>(conv3raw, stats + 44, stats + 164,
                                       w1packfc, ll1_b, ll2_w, ll2_b,
                                       (float*)d_out);
}
